// AdvancedGNN_12790412607664
// MI455X (gfx1250) — matrix operand order repaired and checked by reading, not compiled, not run on hardware
//
#include <hip/hip_runtime.h>
#include <hip/hip_bf16.h>
#include <math.h>

#define DIN   128
#define HID   32
#define NH    4
#define HD    128
#define DOUT  128
#define NC    2
#define NG    256
#define NEG_SLOPE 0.2f
#define BN_EPS    1e-5f

typedef __attribute__((ext_vector_type(16))) _Float16 v16h;
typedef __attribute__((ext_vector_type(8)))  float    v8f;

// ---------- helpers ----------
__device__ __forceinline__ unsigned flip_key(float f) {
  unsigned b = __float_as_uint(f);
  return (b & 0x80000000u) ? ~b : (b | 0x80000000u);
}
__device__ __forceinline__ float unflip_key(unsigned k) {
  unsigned b = (k & 0x80000000u) ? (k & 0x7FFFFFFFu) : ~k;
  return __uint_as_float(b);
}

__global__ void fill_f32(float* __restrict__ p, float v, size_t n) {
  size_t i = (size_t)blockIdx.x * blockDim.x + threadIdx.x;
  if (i < n) p[i] = v;
}

// ---------- pack W[k,m] (f32 row-major) into WMMA-B fragment order (f16) ----------
// Wp[((ct*(k/32)+ks)*32 + lane)*16 + j] = W[(ks*32 + (j&7) + (j>=8)*16 + (lane>=16)*8), ct*16 + lane%16]
__global__ void pack_w_f16(const float* __restrict__ W, _Float16* __restrict__ Wp,
                           int k, int m) {
  int idx = blockIdx.x * blockDim.x + threadIdx.x;
  if (idx >= k * m) return;
  int j    = idx & 15;
  int lane = (idx >> 4) & 31;
  int ks   = (idx >> 9) % (k >> 5);
  int ct   = idx / ((k >> 5) << 9);
  int kb = ks * 32 + (j & 7) + ((j & 8) ? 16 : 0) + ((lane >= 16) ? 8 : 0);
  int nc = ct * 16 + (lane & 15);
  Wp[idx] = (_Float16)W[(size_t)kb * m + nc];
}

// ---------- WMMA GEMM: Y[n,m] = X[n,k] @ W[k,m]; W pre-packed f16 fragments ----------
// one wave per 16x16 tile; branch-free loads (row-clamped), float4 A loads
__global__ __launch_bounds__(32)
void gemm_wmma_f16(const float* __restrict__ X, const _Float16* __restrict__ Wp,
                   float* __restrict__ Y, int n, int k, int m) {
  const int lane = threadIdx.x;
  const int rowBase = blockIdx.x * 16;
  const int colBase = blockIdx.y * 16;
  const int hi  = lane >> 4;
  const int l16 = lane & 15;
  int mrow = rowBase + l16;
  if (mrow >= n) mrow = n - 1;                 // clamp: keep loads unconditional
  const float* xrow = X + (size_t)mrow * k;
  const int ksteps = k >> 5;
  const _Float16* wp = Wp + (((size_t)blockIdx.y * ksteps) << 9) + lane * 16;

  v8f acc = {};
  for (int ks = 0; ks < ksteps; ++ks) {
    // A 16x32 f16 layout: lane holds K = {0..7,16..23} + 8*hi (contiguous float4 runs)
    const float4* xa = (const float4*)(xrow + ks * 32 + hi * 8);
    float4 a0 = xa[0], a1 = xa[1], a2 = xa[4], a3 = xa[5];
    v16h a;
    a[0]=(_Float16)a0.x;  a[1]=(_Float16)a0.y;  a[2]=(_Float16)a0.z;  a[3]=(_Float16)a0.w;
    a[4]=(_Float16)a1.x;  a[5]=(_Float16)a1.y;  a[6]=(_Float16)a1.z;  a[7]=(_Float16)a1.w;
    a[8]=(_Float16)a2.x;  a[9]=(_Float16)a2.y;  a[10]=(_Float16)a2.z; a[11]=(_Float16)a2.w;
    a[12]=(_Float16)a3.x; a[13]=(_Float16)a3.y; a[14]=(_Float16)a3.z; a[15]=(_Float16)a3.w;
    v16h b = *(const v16h*)(wp + ((size_t)ks << 9));
    acc = __builtin_amdgcn_wmma_f32_16x16x32_f16(false, a, false, b,
                                                 (short)0, acc, false, false);
  }
#pragma unroll
  for (int r = 0; r < 8; ++r) {
    int mr = rowBase + r + (hi ? 8 : 0);
    if (mr < n) Y[(size_t)mr * m + colBase + l16] = acc[r];
  }
}

// ---------- attention scalar dots: as/ad [n,NH] ----------
__global__ void alpha_kernel(const float* __restrict__ Hf,
                             const float* __restrict__ a_src,
                             const float* __restrict__ a_dst,
                             float* __restrict__ as, float* __restrict__ ad, int n) {
  int idx = blockIdx.x * blockDim.x + threadIdx.x;
  if (idx >= n * NH) return;
  int node = idx / NH, hh = idx % NH;
  const float* hv = Hf + (size_t)node * HD + hh * HID;
  const float* s = a_src + hh * HID;
  const float* d = a_dst + hh * HID;
  float sa = 0.f, sd = 0.f;
#pragma unroll
  for (int i = 0; i < HID; ++i) { float v = hv[i]; sa += v * s[i]; sd += v * d[i]; }
  as[idx] = sa; ad[idx] = sd;
}

// ---------- edge pass 1: e = leakyrelu(as[src]+ad[dst]); segment max ----------
__global__ void edge_max_kernel(const int* __restrict__ ei, int E, int n,
                                const float* __restrict__ as, const float* __restrict__ ad,
                                float* __restrict__ ebuf, unsigned* __restrict__ mbits) {
  long long idx = (long long)blockIdx.x * blockDim.x + threadIdx.x;
  long long total = (long long)(E + n) * NH;
  if (idx >= total) return;
  int eid = (int)(idx / NH), hh = (int)(idx % NH);
  int src, dst;
  if (eid < E) { src = ei[eid]; dst = ei[E + eid]; } else { src = dst = eid - E; }
  float e = as[(size_t)src * NH + hh] + ad[(size_t)dst * NH + hh];
  e = (e > 0.f) ? e : NEG_SLOPE * e;
  ebuf[idx] = e;
  atomicMax(mbits + (size_t)dst * NH + hh, flip_key(e));
}

// ---------- edge pass 2: segment sum of exp(e - m[dst]) ----------
__global__ void edge_sum_kernel(const int* __restrict__ ei, int E, int n,
                                const float* __restrict__ ebuf,
                                const unsigned* __restrict__ mbits,
                                float* __restrict__ sbuf) {
  long long idx = (long long)blockIdx.x * blockDim.x + threadIdx.x;
  long long total = (long long)(E + n) * NH;
  if (idx >= total) return;
  int eid = (int)(idx / NH), hh = (int)(idx % NH);
  int dst = (eid < E) ? ei[E + eid] : (eid - E);
  float mm = unflip_key(mbits[(size_t)dst * NH + hh]);
  atomicAdd(sbuf + (size_t)dst * NH + hh, __expf(ebuf[idx] - mm));
}

// ---------- edge pass 3: out[dst] += alpha * h[src] ; one wave32 per edge ----------
__global__ void edge_scatter_kernel(const int* __restrict__ ei, int E, int n,
                                    const float* __restrict__ Hf,
                                    const float* __restrict__ ebuf,
                                    const unsigned* __restrict__ mbits,
                                    const float* __restrict__ sbuf,
                                    float* __restrict__ out) {
  int edge = blockIdx.x * (blockDim.x >> 5) + (threadIdx.x >> 5);
  int lane = threadIdx.x & 31;
  int EP = E + n;
  if (edge >= EP) return;
  int src, dst;
  if (edge < E) { src = ei[edge]; dst = ei[E + edge]; } else { src = dst = edge - E; }
  int hh = lane >> 3;                       // 8 lanes (32 floats) per head
  float e  = ebuf[(size_t)edge * NH + hh];
  float mm = unflip_key(mbits[(size_t)dst * NH + hh]);
  float ss = sbuf[(size_t)dst * NH + hh];
  float alpha = __expf(e - mm) / (ss + 1e-16f);
  const float4 hv = *(const float4*)(Hf + (size_t)src * HD + lane * 4);
  float* o = out + (size_t)dst * HD + lane * 4;
  atomicAdd(o + 0, hv.x * alpha);
  atomicAdd(o + 1, hv.y * alpha);
  atomicAdd(o + 2, hv.z * alpha);
  atomicAdd(o + 3, hv.w * alpha);
}

// ---------- fused bias + ReLU (in place) + column sum / sumsq ----------
__global__ void bias_relu_stats(float* __restrict__ Xio, const float* __restrict__ bias,
                                float* __restrict__ csum, float* __restrict__ csq, int n) {
  int g = blockIdx.x * blockDim.x + threadIdx.x;
  int c = g & (HD - 1);
  int nWalk = (gridDim.x * blockDim.x) >> 7;
  int r0 = g >> 7;
  float bc = bias[c];
  float s = 0.f, q = 0.f;
  for (int r = r0; r < n; r += nWalk) {
    float v = Xio[(size_t)r * HD + c] + bc;
    v = v > 0.f ? v : 0.f;
    Xio[(size_t)r * HD + c] = v;
    s += v; q += v * v;
  }
  atomicAdd(csum + c, s);
  atomicAdd(csq + c, q);
}

// ---------- batch-norm apply (in place) ----------
__global__ void bn_apply(float* __restrict__ Xio,
                         const float* __restrict__ csum, const float* __restrict__ csq,
                         const float* __restrict__ gamma, const float* __restrict__ beta,
                         int n) {
  size_t idx = (size_t)blockIdx.x * blockDim.x + threadIdx.x;
  if (idx >= (size_t)n * HD) return;
  int c = (int)(idx & (HD - 1));
  float inv = 1.0f / (float)n;
  float mu  = csum[c] * inv;
  float var = csq[c] * inv - mu * mu;
  float v = (Xio[idx] - mu) * rsqrtf(var + BN_EPS);
  Xio[idx] = gamma[c] * v + beta[c];
}

// ---------- global mean pool ----------
__global__ void pool_kernel(const float* __restrict__ X, const int* __restrict__ batch,
                            float* __restrict__ pooled, float* __restrict__ cnt, int n) {
  size_t idx = (size_t)blockIdx.x * blockDim.x + threadIdx.x;
  if (idx >= (size_t)n * HD) return;
  int node = (int)(idx >> 7);
  int c    = (int)(idx & (HD - 1));
  int gid  = batch[node];
  atomicAdd(pooled + (size_t)gid * HD + c, X[idx]);
  if (c == 0) atomicAdd(cnt + gid, 1.0f);
}

// ---------- scale pooled sums -> means (in place) ----------
__global__ void scale_pool(float* __restrict__ pooled, const float* __restrict__ cnt) {
  int idx = blockIdx.x * blockDim.x + threadIdx.x;
  if (idx >= NG * HD) return;
  int g = idx >> 7;
  pooled[idx] *= 1.0f / fmaxf(cnt[g], 1.0f);
}

// ---------- bias + relu over z (after WMMA L1) ----------
__global__ void bias_relu_vec(float* __restrict__ z, const float* __restrict__ b) {
  int idx = blockIdx.x * blockDim.x + threadIdx.x;
  if (idx >= NG * DOUT) return;
  int j = idx & (DOUT - 1);
  z[idx] = fmaxf(z[idx] + b[j], 0.f);
}

// ---------- final 128 -> 2 projection ----------
__global__ void mlp2_kernel(const float* __restrict__ z,
                            const float* __restrict__ W, const float* __restrict__ b,
                            float* __restrict__ out) {
  int idx = blockIdx.x * blockDim.x + threadIdx.x;
  if (idx >= NG * NC) return;
  int g = idx / NC, c = idx % NC;
  float acc = b[c];
#pragma unroll 8
  for (int j = 0; j < DOUT; ++j)
    acc += z[(size_t)g * DOUT + j] * W[(size_t)j * NC + c];
  out[idx] = acc;
}

// =======================================================================
extern "C" void kernel_launch(void* const* d_in, const int* in_sizes, int n_in,
                              void* d_out, int out_size, void* d_ws, size_t ws_size,
                              hipStream_t stream) {
  (void)n_in; (void)out_size; (void)ws_size;
  const float* x       = (const float*)d_in[0];
  const int*   ei      = (const int*)  d_in[1];
  const int*   batch   = (const int*)  d_in[2];
  const float* W1      = (const float*)d_in[3];
  const float* a1s     = (const float*)d_in[4];
  const float* a1d     = (const float*)d_in[5];
  const float* b1      = (const float*)d_in[6];
  const float* bn1g    = (const float*)d_in[7];
  const float* bn1b    = (const float*)d_in[8];
  const float* W2      = (const float*)d_in[9];
  const float* a2s     = (const float*)d_in[10];
  const float* a2d     = (const float*)d_in[11];
  const float* b2      = (const float*)d_in[12];
  const float* bn2g    = (const float*)d_in[13];
  const float* bn2b    = (const float*)d_in[14];
  const float* L1W     = (const float*)d_in[15];
  const float* L1b     = (const float*)d_in[16];
  const float* L2W     = (const float*)d_in[17];
  const float* L2b     = (const float*)d_in[18];
  float* out = (float*)d_out;

  const int n = in_sizes[0] / DIN;        // 50000
  const int E = in_sizes[1] / 2;          // 1600000
  const int EP = E + n;

  // workspace layout (floats)
  float* ws = (float*)d_ws;
  size_t nHD = (size_t)n * HD;
  float*    hbuf   = ws;                                   // n*HD
  float*    x2     = hbuf + nHD;                           // n*HD (layer1 out / layer2 in)
  float*    o2     = x2 + nHD;                             // n*HD (layer2 out)
  float*    ebuf   = o2 + nHD;                             // EP*NH
  float*    asb    = ebuf + (size_t)EP * NH;               // n*NH
  float*    adb    = asb + (size_t)n * NH;                 // n*NH
  unsigned* mbits  = (unsigned*)(adb + (size_t)n * NH);    // n*NH
  float*    sbuf   = (float*)mbits + (size_t)n * NH;       // n*NH
  float*    csum   = sbuf + (size_t)n * NH;                // HD
  float*    csq    = csum + HD;                            // HD
  float*    pooled = csq + HD;                             // NG*HD
  float*    cnt    = pooled + (size_t)NG * HD;             // NG
  float*    zbuf   = cnt + NG;                             // NG*DOUT
  size_t wpOff = (size_t)(zbuf + NG * DOUT - ws);
  wpOff = (wpOff + 15) & ~(size_t)15;                      // 64B align
  _Float16* Wp = (_Float16*)(ws + wpOff);                  // k*m halfs (<=16384)

  const int TB = 256;
  dim3 gemmGrid((n + 15) / 16, HD / 16);
  long long ehTotal = (long long)EP * NH;
  int ehBlocks   = (int)((ehTotal + TB - 1) / TB);
  int nhBlocks   = (int)(((size_t)n * NH + TB - 1) / TB);
  int nHDBlocks  = (int)((nHD + TB - 1) / TB);
  int scatBlocks = (EP + 7) / 8;                           // 8 edges (waves) / block
  int packBlocks = (DIN * HD + TB - 1) / TB;

  for (int layer = 0; layer < 2; ++layer) {
    const float* xin  = (layer == 0) ? x   : x2;
    const float* Wl   = (layer == 0) ? W1  : W2;
    const float* asl  = (layer == 0) ? a1s : a2s;
    const float* adl  = (layer == 0) ? a1d : a2d;
    const float* bl   = (layer == 0) ? b1  : b2;
    const float* bng  = (layer == 0) ? bn1g : bn2g;
    const float* bnb  = (layer == 0) ? bn1b : bn2b;
    float*       olay = (layer == 0) ? x2  : o2;
    const int    kin  = (layer == 0) ? DIN : HD;

    // h = xin @ W  (pack W -> WMMA)
    pack_w_f16<<<packBlocks, TB, 0, stream>>>(Wl, Wp, kin, HD);
    gemm_wmma_f16<<<gemmGrid, 32, 0, stream>>>(xin, Wp, hbuf, n, kin, HD);
    // attention scalars
    alpha_kernel<<<nhBlocks, TB, 0, stream>>>(hbuf, asl, adl, asb, adb, n);
    // softmax over incoming edges
    fill_f32<<<nhBlocks, TB, 0, stream>>>((float*)mbits, 0.0f, (size_t)n * NH);
    edge_max_kernel<<<ehBlocks, TB, 0, stream>>>(ei, E, n, asb, adb, ebuf, mbits);
    fill_f32<<<nhBlocks, TB, 0, stream>>>(sbuf, 0.0f, (size_t)n * NH);
    edge_sum_kernel<<<ehBlocks, TB, 0, stream>>>(ei, E, n, ebuf, mbits, sbuf);
    // weighted scatter-add
    fill_f32<<<nHDBlocks, TB, 0, stream>>>(olay, 0.0f, nHD);
    edge_scatter_kernel<<<scatBlocks, TB, 0, stream>>>(ei, E, n, hbuf, ebuf, mbits, sbuf, olay);
    // bias + relu + BN
    fill_f32<<<1, TB, 0, stream>>>(csum, 0.0f, 2 * HD);    // csum & csq contiguous
    bias_relu_stats<<<512, TB, 0, stream>>>(olay, bl, csum, csq, n);
    bn_apply<<<nHDBlocks, TB, 0, stream>>>(olay, csum, csq, bng, bnb, n);
  }

  // global mean pool
  fill_f32<<<(NG * HD + NG + TB - 1) / TB, TB, 0, stream>>>(pooled, 0.0f, (size_t)NG * HD + NG);
  pool_kernel<<<nHDBlocks, TB, 0, stream>>>(o2, batch, pooled, cnt, n);
  scale_pool<<<(NG * HD + TB - 1) / TB, TB, 0, stream>>>(pooled, cnt);
  // MLP layer 1 via WMMA: z = relu(pooled @ L1W + L1b)
  pack_w_f16<<<packBlocks, TB, 0, stream>>>(L1W, Wp, HD, DOUT);
  dim3 mlpGrid(NG / 16, DOUT / 16);
  gemm_wmma_f16<<<mlpGrid, 32, 0, stream>>>(pooled, Wp, zbuf, NG, HD, DOUT);
  bias_relu_vec<<<(NG * DOUT + TB - 1) / TB, TB, 0, stream>>>(zbuf, L1b);
  // final projection
  mlp2_kernel<<<(NG * NC + TB - 1) / TB, TB, 0, stream>>>(zbuf, L2W, L2b, out);
}
